// DSDM_39702677684486
// MI455X (gfx1250) — matrix operand order repaired and checked by reading, not compiled, not run on hardware
//
#include <hip/hip_runtime.h>

// ---------------------------------------------------------------------------
// Fused cosine-sim softmin retrieval ("attention") for MI455X (gfx1250).
//   Q=normalize(query)[64x512], logits = 10 * (Q . a_n / max(||a_n||,eps)),
//   out = softmax(logits) @ addresses.
// Memory-bound: addresses (410MB) streamed exactly once (non-temporal),
// both GEMMs on v_wmma_f32_16x16x32_bf16 (f32 accumulation).
// Register double-buffering hides DRAM latency behind the WMMA phases.
// Split-N reduction: per-block partials in workspace (preferred) or f32
// global atomics (fallback if ws_size is small).
// ---------------------------------------------------------------------------

typedef __attribute__((ext_vector_type(16))) __bf16        v16bf;
typedef __attribute__((ext_vector_type(8)))  float         v8f;
typedef __attribute__((ext_vector_type(4)))  float         f32x4;
typedef __attribute__((ext_vector_type(4)))  unsigned int  u32x4;

union Frag {            // 32 bytes: one bf16 WMMA A/B operand per lane
  v16bf v;
  u32x4 q[2];
};

__device__ __forceinline__ unsigned short f2bf(float x) {   // f32 -> bf16 RNE
  unsigned int u = __float_as_uint(x);
  unsigned int r = u + 0x7FFFu + ((u >> 16) & 1u);
  return (unsigned short)(r >> 16);
}
__device__ __forceinline__ float bf2f(unsigned short h) {
  return __uint_as_float(((unsigned int)h) << 16);
}

constexpr int D  = 512;   // feature dim
constexpr int B  = 64;    // queries
constexpr int TN = 32;    // addresses per tile (K of GEMM2, N of GEMM1)
constexpr int QS = D + 8; // padded LDS row stride (bf16) -> conflict-free frags
constexpr int TS = TN + 8;
constexpr int PS = TN + 8;
constexpr int PART_STRIDE = B * D + B;   // floats per per-block partial

// LDS layout in bf16 (ushort) elements
constexpr int OFF_QH = 0;                 // Qn   : B  x QS
constexpr int OFF_AH = OFF_QH + B  * QS;  // tile : TN x QS   (row-major, GEMM1 B)
constexpr int OFF_TH = OFF_AH + TN * QS;  // tileT: D  x TS   (d-major,  GEMM2 B)
constexpr int OFF_PM = OFF_TH + D  * TS;  // P    : B  x PS   (weights,  GEMM2 A)
constexpr int USHORT_TOTAL = OFF_PM + B * PS;
constexpr int SMEM_BYTES = USHORT_TOTAL * 2 + 96 * 4;   // + ssq/scl/msk (~146KB)

// ---------------------------------------------------------------------------
__global__ void dsdm_prep(const float* __restrict__ q,
                          unsigned short* __restrict__ qh_g,
                          float* __restrict__ num_g,
                          float* __restrict__ den_g)
{
  __shared__ float ssum;
  const int b = blockIdx.x, tid = threadIdx.x;
  if (tid == 0) ssum = 0.0f;
  __syncthreads();
  const float x0 = q[b * D + tid];
  const float x1 = q[b * D + 256 + tid];
  atomicAdd(&ssum, x0 * x0 + x1 * x1);
  __syncthreads();
  const float inv = 1.0f / fmaxf(sqrtf(ssum), 1e-8f);   // torch eps clamp
  qh_g[b * D + tid]       = f2bf(x0 * inv);
  qh_g[b * D + 256 + tid] = f2bf(x1 * inv);
  num_g[b * D + tid]       = 0.0f;                      // atomic-mode init
  num_g[b * D + 256 + tid] = 0.0f;
  if (tid == 0) den_g[b] = 0.0f;
}

// ---------------------------------------------------------------------------
__global__ __launch_bounds__(256)
void dsdm_main(const float* __restrict__ addr,
               const unsigned short* __restrict__ qh_g,
               float* __restrict__ num_g,
               float* __restrict__ den_g,
               float* __restrict__ part,   // non-null => partial mode
               int N)
{
  extern __shared__ char smem_raw[];
  unsigned short* sm = (unsigned short*)smem_raw;
  unsigned short* Qh = sm + OFF_QH;
  unsigned short* Ah = sm + OFF_AH;
  unsigned short* Th = sm + OFF_TH;
  unsigned short* Pm = sm + OFF_PM;
  float* ssq = (float*)(sm + USHORT_TOTAL);   // 32 (ssq) + 32 (scl) + 32 (msk)
  float* scl = ssq + 32;
  float* msk = scl + 32;

  const int tid = threadIdx.x;
  const int l   = tid & 31;       // lane (wave32)
  const int w   = tid >> 5;       // 8 waves
  const int m   = w & 3;          // query mtile (16 rows)
  const int nh  = w >> 2;         // GEMM1 addr-half / GEMM2 d-half

  // stage normalized bf16 queries into LDS (padded rows), b128 chunks
  for (int i = tid; i < (B * D) / 8; i += 256) {
    const int row = i >> 6;            // 64 8-elem chunks per row
    const int col = (i & 63) << 3;
    *(u32x4*)(Qh + row * QS + col) = ((const u32x4*)qh_g)[i];
  }
  if (tid < 32) ssq[tid] = 0.0f;
  __syncthreads();

  // persistent accumulators: 16 C-tiles (128 VGPRs) + row-sum partials
  v8f acc2[16];
  #pragma unroll
  for (int i = 0; i < 16; ++i)
    #pragma unroll
    for (int r = 0; r < 8; ++r) acc2[i][r] = 0.0f;
  float den_acc[8];
  #pragma unroll
  for (int r = 0; r < 8; ++r) den_acc[r] = 0.0f;

  const int NT = (N + TN - 1) / TN;
  const int r8 = tid >> 3, s8 = tid & 7;
  const int d0 = s8 * 64;

  // register double-buffer: 16 x f32x4 of the tile this thread owns
  f32x4 xbuf[16];
  auto load_tile = [&](int tt) {
    const int n = tt * TN + r8;
    const bool v = (tt < NT) && (n < N);
    const float* arow = addr + (size_t)(v ? n : 0) * D;
    #pragma unroll
    for (int j = 0; j < 16; ++j) {
      f32x4 z = {0.0f, 0.0f, 0.0f, 0.0f};
      if (v) z = __builtin_nontemporal_load((const f32x4*)(arow + d0 + j * 4));
      xbuf[j] = z;
    }
  };
  load_tile(blockIdx.x);   // prologue

  for (int t = blockIdx.x; t < NT; t += gridDim.x) {
    const int n0 = t * TN;
    __syncthreads();  // previous iteration's GEMM reads complete

    // ---- convert prefetched tile to bf16 in LDS; per-row sumsq ----
    {
      float sq = 0.0f;
      #pragma unroll
      for (int j = 0; j < 16; ++j) {
        const f32x4 x = xbuf[j];
        sq += x.x * x.x + x.y * x.y + x.z * x.z + x.w * x.w;
        unsigned long long pk =
            (unsigned long long)f2bf(x.x)         | ((unsigned long long)f2bf(x.y) << 16)
          | ((unsigned long long)f2bf(x.z) << 32) | ((unsigned long long)f2bf(x.w) << 48);
        *(unsigned long long*)(Ah + r8 * QS + d0 + j * 4) = pk;
      }
      atomicAdd(&ssq[r8], sq);
    }
    // issue next tile's loads now; they fly through transpose + both GEMMs
    load_tile(t + (int)gridDim.x);
    __syncthreads();

    // ---- transpose tile into Th (d-major) for GEMM2 B operand ----
    for (int dd = tid; dd < D; dd += 256) {
      alignas(16) unsigned short col[TN];
      #pragma unroll
      for (int rr = 0; rr < TN; ++rr) col[rr] = Ah[rr * QS + dd];
      u32x4* dst = (u32x4*)(Th + dd * TS);
      dst[0] = *(const u32x4*)(col + 0);
      dst[1] = *(const u32x4*)(col + 8);
      dst[2] = *(const u32x4*)(col + 16);
      dst[3] = *(const u32x4*)(col + 24);
    }
    if (tid < 32) {
      const float nrm = sqrtf(ssq[tid]);
      scl[tid] = 14.4269504088896341f / fmaxf(nrm, 1e-8f); // 10*log2e/||a||
      msk[tid] = ((n0 + tid) < N) ? 1.0f : 0.0f;           // ragged tail
      ssq[tid] = 0.0f;                                     // for next tile
    }
    __syncthreads();

    // ---- GEMM1: S[64x32] = Qn @ tile^T on the tensor pipe ----
    {
      v8f s;
      #pragma unroll
      for (int r = 0; r < 8; ++r) s[r] = 0.0f;
      const int arw   = m * 16 + (l & 15);   // A row (query)
      const int abase = (l >> 4) << 3;       // A K-chunk base: 0 / 8
      const int brw   = nh * 16 + (l & 15);  // B col (address)
      const int bbase = (l >> 4) << 3;       // B K-chunk base: 0 / 8
      #pragma unroll
      for (int kt = 0; kt < D / 32; ++kt) {
        Frag fa, fb;
        const unsigned short* ap = Qh + arw * QS + kt * 32 + abase;
        fa.q[0] = *(const u32x4*)(ap);
        fa.q[1] = *(const u32x4*)(ap + 16);
        const unsigned short* bp = Ah + brw * QS + kt * 32 + bbase;
        fb.q[0] = *(const u32x4*)(bp);
        fb.q[1] = *(const u32x4*)(bp + 16);
        s = __builtin_amdgcn_wmma_f32_16x16x32_bf16(false, fa.v, false, fb.v,
                                                    (short)0, s, false, false);
      }
      // w = exp2(scale * dot); logits bounded in [-10,10] -> no running max
      const int col = nh * 16 + (l & 15);
      const float sc = scl[col];
      const float mk = msk[col];
      const int rbase = m * 16 + ((l >> 4) << 3);
      #pragma unroll
      for (int r = 0; r < 8; ++r) {
        const float wgt = mk * exp2f(s[r] * sc);
        const unsigned short wb = f2bf(wgt);
        Pm[(rbase + r) * PS + col] = wb;
        den_acc[r] += bf2f(wb);   // same rounded weight as numerator GEMM
      }
    }
    __syncthreads();

    // ---- GEMM2: O[64x512] += P[64x32] @ tile[32x512] ----
    {
      Frag fp;
      const int prw = m * 16 + (l & 15);
      const unsigned short* pp = Pm + prw * PS + ((l >> 4) << 3);
      fp.q[0] = *(const u32x4*)(pp);
      fp.q[1] = *(const u32x4*)(pp + 16);
      const int kb = (l >> 4) << 3;
      #pragma unroll
      for (int dt = 0; dt < 16; ++dt) {
        const int d = nh * 256 + dt * 16 + (l & 15);
        Frag fv;
        const unsigned short* vp = Th + d * TS + kb;
        fv.q[0] = *(const u32x4*)(vp);
        fv.q[1] = *(const u32x4*)(vp + 16);
        acc2[dt] = __builtin_amdgcn_wmma_f32_16x16x32_bf16(false, fp.v, false, fv.v,
                                                           (short)0, acc2[dt],
                                                           false, false);
      }
    }
  }

  // ---- epilogue: emit split-N partials ----
  const int rbase = m * 16 + ((l >> 4) << 3);
  if (part != nullptr) {
    // partial mode: plain stores into this block's private region
    float* pn = part + (size_t)blockIdx.x * PART_STRIDE;
    float* pd = pn + B * D;
    #pragma unroll
    for (int dt = 0; dt < 16; ++dt) {
      const int d = nh * 256 + dt * 16 + (l & 15);
      #pragma unroll
      for (int r = 0; r < 8; ++r)
        pn[(rbase + r) * D + d] = acc2[dt][r];
    }
    __syncthreads();               // ssq/scl free for reuse as den staging
    if (tid < B) ssq[tid] = 0.0f;  // 64 floats: ssq(32)+scl(32) contiguous
    __syncthreads();
    #pragma unroll
    for (int r = 0; r < 8; ++r) atomicAdd(&ssq[rbase + r], den_acc[r]);
    __syncthreads();
    if (tid < B) pd[tid] = ssq[tid];
  } else {
    // fallback: f32 global atomics into zero-initialized accumulators
    #pragma unroll
    for (int r = 0; r < 8; ++r)
      atomicAdd(&den_g[rbase + r], den_acc[r]);
    #pragma unroll
    for (int dt = 0; dt < 16; ++dt) {
      const int d = nh * 256 + dt * 16 + (l & 15);
      #pragma unroll
      for (int r = 0; r < 8; ++r)
        atomicAdd(&num_g[(rbase + r) * D + d], acc2[dt][r]);
    }
  }
}

// ---------------------------------------------------------------------------
__global__ void dsdm_reduce(const float* __restrict__ part, int npart,
                            float* __restrict__ out)
{
  __shared__ float sden;
  const int b = blockIdx.x, tid = threadIdx.x;
  if (tid == 0) sden = 0.0f;
  __syncthreads();
  float dsum = 0.0f;
  for (int g = tid; g < npart; g += 256)
    dsum += part[(size_t)g * PART_STRIDE + B * D + b];
  atomicAdd(&sden, dsum);
  float s0 = 0.0f, s1 = 0.0f;
  for (int g = 0; g < npart; ++g) {
    const float* p = part + (size_t)g * PART_STRIDE + b * D;
    s0 += p[tid];
    s1 += p[tid + 256];
  }
  __syncthreads();
  const float inv = 1.0f / sden;
  out[b * D + tid]       = s0 * inv;
  out[b * D + 256 + tid] = s1 * inv;
}

// ---------------------------------------------------------------------------
__global__ void dsdm_finalize(const float* __restrict__ num_g,
                              const float* __restrict__ den_g,
                              float* __restrict__ out)
{
  const int b = blockIdx.x, tid = threadIdx.x;
  const float inv = 1.0f / den_g[b];
  out[b * D + tid]       = num_g[b * D + tid] * inv;
  out[b * D + 256 + tid] = num_g[b * D + 256 + tid] * inv;
}

// ---------------------------------------------------------------------------
extern "C" void kernel_launch(void* const* d_in, const int* in_sizes, int n_in,
                              void* d_out, int out_size, void* d_ws, size_t ws_size,
                              hipStream_t stream)
{
  const float* query     = (const float*)d_in[0];
  const float* addresses = (const float*)d_in[1];
  const int N = in_sizes[1] / D;    // 200000

  // ws layout: bf16 Qn (64KB) | f32 num (128KB) | f32 den (256B) | partials
  unsigned short* qh_g = (unsigned short*)d_ws;
  float* num_g = (float*)((char*)d_ws + (size_t)B * D * sizeof(unsigned short));
  float* den_g = num_g + B * D;
  float* part  = den_g + B;
  const size_t fixed_bytes = (size_t)((char*)part - (char*)d_ws);

  const int NT = (N + TN - 1) / TN;
  const size_t avail = ws_size > fixed_bytes ? ws_size - fixed_bytes : 0;
  size_t pcap = avail / ((size_t)PART_STRIDE * sizeof(float));
  int pblocks = pcap > 384 ? 384 : (int)pcap;

  dsdm_prep<<<B, 256, 0, stream>>>(query, qh_g, num_g, den_g);

  if (pblocks >= 64) {
    // preferred: deterministic per-block partials, no global atomics
    const int grid = pblocks < NT ? pblocks : NT;
    dsdm_main<<<grid, 256, SMEM_BYTES, stream>>>(addresses, qh_g,
                                                 num_g, den_g, part, N);
    dsdm_reduce<<<B, 256, 0, stream>>>(part, grid, (float*)d_out);
  } else {
    // fallback: atomic accumulation with a modest grid to bound atomic count
    const int grid = NT < 128 ? NT : 128;
    dsdm_main<<<grid, 256, SMEM_BYTES, stream>>>(addresses, qh_g,
                                                 num_g, den_g, nullptr, N);
    dsdm_finalize<<<B, 256, 0, stream>>>(num_g, den_g, (float*)d_out);
  }
}
